// HGCN_39771397161685
// MI455X (gfx1250) — matrix operand order repaired and checked by reading, not compiled, not run on hardware
//
#include <hip/hip_runtime.h>

#define NROWS 8192
#define DDIM  64

#define EPS      1e-7f
#define MIN_NORM 1e-15f
#define MAX_NORM 1e6f
#define SLOPE    0.01f

typedef __attribute__((ext_vector_type(8)))  float  v8f;
typedef __attribute__((ext_vector_type(4)))  __bf16 v4bf;
typedef __attribute__((ext_vector_type(8)))  __bf16 v8bf;
typedef __attribute__((ext_vector_type(16))) __bf16 v16bf;
typedef int v4i __attribute__((vector_size(16)));   // 128-bit payload type

// CDNA5 async global->LDS path (gfx1250). Guarded so the host pass and older
// toolchains fall back to synchronous copies with the identical LDS layout.
#if defined(__has_builtin)
#if __has_builtin(__builtin_amdgcn_global_load_async_to_lds_b128) && \
    __has_builtin(__builtin_amdgcn_s_wait_asynccnt)
#define USE_ASYNC 1
#endif
#endif
#ifndef USE_ASYNC
#define USE_ASYNC 0
#endif

__device__ __forceinline__ float clip15(float x){ return fminf(15.f, fmaxf(-15.f, x)); }
__device__ __forceinline__ float dsinh(float x){ return sinhf(clip15(x)); }
__device__ __forceinline__ float dcosh(float x){ return coshf(clip15(x)); }
__device__ __forceinline__ float darcosh(float x){
  x = fmaxf(x, 1.f + EPS);
  return logf(x + sqrtf(x * x - 1.f));
}
// sum across the 8-lane group that owns one row (lanes are 8-aligned in the wave)
__device__ __forceinline__ float grp8sum(float v){
  v += __shfl_xor(v, 1, 32);
  v += __shfl_xor(v, 2, 32);
  v += __shfl_xor(v, 4, 32);
  return v;
}

// ---------------------------------------------------------------------------
// Kernel 1: ub = logmap0(proj(expmap0(proj_tan0(bias))))   (64 floats)
// ---------------------------------------------------------------------------
__global__ void bias_prep(const float* __restrict__ bias, float* __restrict__ ub){
  __shared__ float red[64];
  int t = threadIdx.x;               // 0..63
  float b = (t == 0) ? 0.f : bias[t];
  red[t] = b * b;                    // t==0 contributes 0
  __syncthreads();
  for (int s = 32; s > 0; s >>= 1){
    if (t < s) red[t] += red[t + s];
    __syncthreads();
  }
  float yn2 = fmaxf(red[0], 1e-30f);
  float yn  = fmaxf(sqrtf(yn2), MIN_NORM);
  float sc  = dsinh(yn) / yn;
  float rn  = sc * yn;                               // ||r||
  float x0  = sqrtf(fmaxf(1.f + rn * rn, EPS));
  float g   = darcosh(fmaxf(x0, 1.f + EPS)) / fmaxf(rn, MIN_NORM);
  ub[t] = (t == 0) ? 0.f : g * sc * b;
}

// ---------------------------------------------------------------------------
// Kernel 2: fused  M1 = logmap0(proj(expmap0(proj_tan0(x))))
//                  Z  = M1 @ W^T            (WMMA bf16, K=64)
//                  res= proj(expmap0(Z)); mobius_add(res, hyp_bias)
//                  L2t= logmap0(proj(res))  stored bf16 TRANSPOSED (64 x 8192)
// 16 rows / block, 128 threads (4 waves = 4 N-tiles of 16)
// ---------------------------------------------------------------------------
__global__ void pre_kernel(const float* __restrict__ x,
                           const float* __restrict__ W,
                           const float* __restrict__ ub,
                           __bf16* __restrict__ L2t){
  __shared__ __align__(64) __bf16 As[16][64];    // A tile (M1), pitch 128B
  __shared__ __align__(64) __bf16 Wt[64][80];    // Bt[n][k] = W[n][k], pitch 160B
  __shared__ float Zs[16][68];
  __shared__ float ubs[64];

  const int tid  = threadIdx.x;
  const int wave = tid >> 5;
  const int lane = tid & 31;
  const int ln16 = lane & 15;
  const int kh   = lane >> 4;
  const int row  = tid >> 3;          // 0..15 (8 lanes per row)
  const int sub  = tid & 7;
  const int r0   = blockIdx.x * 16;

  // ---- stage A: per-row logmap0(proj(expmap0(proj_tan0(x_row)))) -> bf16 ----
  float v[8];
  {
    const float* xr = x + (size_t)(r0 + row) * DDIM + sub * 8;
    float4 a = *(const float4*)xr;
    float4 b = *(const float4*)(xr + 4);
    v[0]=a.x; v[1]=a.y; v[2]=a.z; v[3]=a.w; v[4]=b.x; v[5]=b.y; v[6]=b.z; v[7]=b.w;
    float yn2 = 0.f;
    #pragma unroll
    for (int i = 0; i < 8; ++i){
      bool c0 = (sub == 0 && i == 0);
      yn2 += c0 ? 0.f : v[i] * v[i];
    }
    yn2 = grp8sum(yn2);
    float yn = fmaxf(sqrtf(fmaxf(yn2, 1e-30f)), MIN_NORM);
    float sc = dsinh(yn) / yn;
    float rn = sc * yn;
    float x0 = sqrtf(fmaxf(1.f + rn * rn, EPS));
    float g  = darcosh(fmaxf(x0, 1.f + EPS)) / fmaxf(rn, MIN_NORM);
    #pragma unroll
    for (int i = 0; i < 8; ++i){
      bool c0 = (sub == 0 && i == 0);
      As[row][sub * 8 + i] = (__bf16)(c0 ? 0.f : g * sc * v[i]);
    }
  }
  // ---- stage weight (Bt == W row-major) and ub ----
  if (tid < 64) ubs[tid] = ub[tid];
  for (int idx = tid; idx < 64 * 64; idx += 128)
    Wt[idx >> 6][idx & 63] = (__bf16)W[idx];
  __syncthreads();

  // ---- WMMA: wave handles cols 16*wave .. +15, K = 64 in two slices of 32 ----
  v8f acc = {0.f,0.f,0.f,0.f,0.f,0.f,0.f,0.f};
  #pragma unroll
  for (int ks = 0; ks < 64; ks += 32){
    v8bf lo = *(const v8bf*)&As[ln16][ks + 16 * kh];
    v8bf hi = *(const v8bf*)&As[ln16][ks + 16 * kh + 8];
    v16bf Am = __builtin_shufflevector(lo, hi, 0,1,2,3,4,5,6,7,8,9,10,11,12,13,14,15);
    v16bf Bm = *(const v16bf*)&Wt[wave * 16 + ln16][ks + 16 * kh];
    acc = __builtin_amdgcn_wmma_f32_16x16x32_bf16(false, Am, false, Bm,
                                                  (short)0, acc, false, false);
  }
  #pragma unroll
  for (int j = 0; j < 8; ++j)
    Zs[j + 8 * kh][wave * 16 + ln16] = acc[j];
  __syncthreads();

  // ---- epilogue per row: proj(expmap0(Z)), mobius_add(.,hb), logmap0 ----
  {
    float z[8];
    #pragma unroll
    for (int i = 0; i < 8; ++i) z[i] = Zs[row][sub * 8 + i];

    float yn2 = 0.f;
    #pragma unroll
    for (int i = 0; i < 8; ++i){
      bool c0 = (sub == 0 && i == 0);
      yn2 += c0 ? 0.f : z[i] * z[i];
    }
    yn2 = grp8sum(yn2);
    float yn = fmaxf(sqrtf(fmaxf(yn2, 1e-30f)), MIN_NORM);
    float sc = dsinh(yn) / yn;
    float rn = sc * yn;
    float x0 = sqrtf(fmaxf(1.f + rn * rn, EPS));
    float res[8];
    #pragma unroll
    for (int i = 0; i < 8; ++i){
      bool c0 = (sub == 0 && i == 0);
      res[i] = c0 ? x0 : sc * z[i];
    }
    // ptransp0(res, ub)
    float rnb = fmaxf(rn, MIN_NORM);
    float alpha = 0.f;
    #pragma unroll
    for (int i = 0; i < 8; ++i){
      bool c0 = (sub == 0 && i == 0);
      alpha += c0 ? 0.f : (res[i] / rnb) * ubs[sub * 8 + i];
    }
    alpha = grp8sum(alpha);
    float w[8];
    #pragma unroll
    for (int i = 0; i < 8; ++i){
      bool c0 = (sub == 0 && i == 0);
      w[i] = c0 ? (alpha * rnb)
                : ubs[sub * 8 + i] - alpha * (1.f - x0) * (res[i] / rnb);
    }
    // proj_tan(w, res)
    float ux = 0.f;
    #pragma unroll
    for (int i = 0; i < 8; ++i){
      bool c0 = (sub == 0 && i == 0);
      ux += c0 ? 0.f : res[i] * w[i];
    }
    ux = grp8sum(ux);
    float v0 = ux / fmaxf(x0, EPS);
    float vv[8];
    #pragma unroll
    for (int i = 0; i < 8; ++i){
      bool c0 = (sub == 0 && i == 0);
      vv[i] = c0 ? v0 : w[i];
    }
    // expmap(vv, res)
    float sq = 0.f;
    #pragma unroll
    for (int i = 0; i < 8; ++i) sq += vv[i] * vv[i];
    sq = grp8sum(sq);
    float md = sq - 2.f * v0 * v0;
    float nu = fminf(sqrtf(fmaxf(md, EPS)), MAX_NORM);
    float th = fmaxf(nu, MIN_NORM);
    float ch = dcosh(th), sh = dsinh(th) / th;
    float e[8];
    #pragma unroll
    for (int i = 0; i < 8; ++i) e[i] = ch * res[i] + sh * vv[i];
    // proj + logmap0
    float en2 = 0.f;
    #pragma unroll
    for (int i = 0; i < 8; ++i){
      bool c0 = (sub == 0 && i == 0);
      en2 += c0 ? 0.f : e[i] * e[i];
    }
    en2 = grp8sum(en2);
    float e0 = sqrtf(fmaxf(1.f + en2, EPS));
    float lnrm = fmaxf(sqrtf(fmaxf(en2, 1e-30f)), MIN_NORM);
    float gg = darcosh(fmaxf(e0, 1.f + EPS)) / lnrm;
    // write bf16 transposed panel: L2t[n][k],  k = global row index
    #pragma unroll
    for (int i = 0; i < 8; ++i){
      bool c0 = (sub == 0 && i == 0);
      int n = sub * 8 + i;
      L2t[(size_t)n * NROWS + (r0 + row)] = (__bf16)(c0 ? 0.f : gg * e[i]);
    }
  }
}

// ---------------------------------------------------------------------------
// Kernel 3: H = adj @ L2 (8192x8192 @ 8192x64, bf16 WMMA, K-chunks of 128,
// double-buffered LDS). A staged sync (needs f32->bf16 cvt); B staged with
// GLOBAL_LOAD_ASYNC_TO_LDS_B128 from the pre-transposed bf16 panel.
// Fused epilogue:
// out = proj(expmap0(proj_tan0(leaky_relu(logmap0(proj(expmap0(H)))))))
// ---------------------------------------------------------------------------
__global__ void agg_kernel(const float* __restrict__ adj,
                           const __bf16* __restrict__ L2t,
                           float* __restrict__ out){
  __shared__ __align__(64) __bf16 As[2][16][128];   // pitch 256B
  __shared__ __align__(64) __bf16 Bt[2][64][128];   // Bt[n][k], pitch 256B
  __shared__ float Hs[16][68];

  const int tid  = threadIdx.x;
  const int wave = tid >> 5;
  const int lane = tid & 31;
  const int ln16 = lane & 15;
  const int kh   = lane >> 4;
  const int row  = tid >> 3;
  const int sub  = tid & 7;
  const int r0   = blockIdx.x * 16;
  const int NCH  = NROWS / 128;      // 64 chunks

  v8f acc = {0.f,0.f,0.f,0.f,0.f,0.f,0.f,0.f};

  // ---- B stage: 64 rows x 128 k of bf16 = 16KB = 1024 x 16B granules ----
  auto stageB = [&](int c, int nb){
    const int k0 = c * 128;
    #pragma unroll
    for (int i = 0; i < 8; ++i){
      int idx = tid + i * 128;              // 0..1023
      int n = idx >> 4;                     // 16 granules per n-row
      int g = idx & 15;
      const __bf16* gp = L2t + (size_t)n * NROWS + k0 + g * 8;
      __bf16* lp = &Bt[nb][n][g * 8];
#if USE_ASYNC
      __builtin_amdgcn_global_load_async_to_lds_b128(
          (__attribute__((address_space(1))) v4i*)gp,
          (__attribute__((address_space(3))) v4i*)lp, 0, 0);
#else
      *(uint4*)lp = *(const uint4*)gp;
#endif
    }
  };
  // ---- A stage: 16 rows x 128 k f32 -> bf16 (conversion needs VGPR pass) ----
  auto stageA = [&](int c, int nb){
    const int k0 = c * 128;
    #pragma unroll
    for (int i = 0; i < 4; ++i){
      int idx = tid + i * 128;              // 512 float4
      int m = idx >> 5, k4 = (idx & 31) * 4;
      float4 f = *(const float4*)&adj[(size_t)(r0 + m) * NROWS + k0 + k4];
      v4bf p = {(__bf16)f.x, (__bf16)f.y, (__bf16)f.z, (__bf16)f.w};
      *(v4bf*)&As[nb][m][k4] = p;
    }
  };

  stageB(0, 0);
  stageA(0, 0);
#if USE_ASYNC
  __builtin_amdgcn_s_wait_asynccnt(0);
#endif
  __syncthreads();

  for (int c = 0; c < NCH; ++c){
    const int buf = c & 1;
    if (c + 1 < NCH){
      stageB(c + 1, buf ^ 1);               // async: fire & forget, wait below
      stageA(c + 1, buf ^ 1);
    }
    // prefetch adj chunk c+2 (global_prefetch_b8): 128 thr x 64B = full 8KB chunk
    if (c + 2 < NCH){
      const float* pf = &adj[(size_t)(r0 + (tid >> 3)) * NROWS + (c + 2) * 128 + (tid & 7) * 16];
      __builtin_prefetch(pf, 0, 1);
    }
    // ---- compute: 4 WMMA per wave on chunk `buf` ----
    #pragma unroll
    for (int ks = 0; ks < 128; ks += 32){
      v8bf lo = *(const v8bf*)&As[buf][ln16][ks + 16 * kh];
      v8bf hi = *(const v8bf*)&As[buf][ln16][ks + 16 * kh + 8];
      v16bf Am = __builtin_shufflevector(lo, hi, 0,1,2,3,4,5,6,7,8,9,10,11,12,13,14,15);
      v16bf Bm = *(const v16bf*)&Bt[buf][wave * 16 + ln16][ks + 16 * kh];
      acc = __builtin_amdgcn_wmma_f32_16x16x32_bf16(false, Am, false, Bm,
                                                    (short)0, acc, false, false);
    }
#if USE_ASYNC
    __builtin_amdgcn_s_wait_asynccnt(0);    // own async stores done before signal
#endif
    __syncthreads();
  }

  // ---- scatter accumulators ----
  #pragma unroll
  for (int j = 0; j < 8; ++j)
    Hs[j + 8 * kh][wave * 16 + ln16] = acc[j];
  __syncthreads();

  // ---- fused row epilogue ----
  {
    float h[8];
    #pragma unroll
    for (int i = 0; i < 8; ++i) h[i] = Hs[row][sub * 8 + i];

    float yn2 = 0.f;
    #pragma unroll
    for (int i = 0; i < 8; ++i){
      bool c0 = (sub == 0 && i == 0);
      yn2 += c0 ? 0.f : h[i] * h[i];
    }
    yn2 = grp8sum(yn2);
    float yn = fmaxf(sqrtf(fmaxf(yn2, 1e-30f)), MIN_NORM);
    float s1 = dsinh(yn) / yn;
    float rn = s1 * yn;
    float x0 = sqrtf(fmaxf(1.f + rn * rn, EPS));
    float g  = darcosh(fmaxf(x0, 1.f + EPS)) / fmaxf(rn, MIN_NORM);
    float t[8];
    #pragma unroll
    for (int i = 0; i < 8; ++i){
      bool c0 = (sub == 0 && i == 0);
      float q = c0 ? 0.f : g * s1 * h[i];
      t[i] = (q > 0.f) ? q : SLOPE * q;
    }
    float tn2 = 0.f;
    #pragma unroll
    for (int i = 0; i < 8; ++i){
      bool c0 = (sub == 0 && i == 0);
      tn2 += c0 ? 0.f : t[i] * t[i];
    }
    tn2 = grp8sum(tn2);
    float tn = fmaxf(sqrtf(fmaxf(tn2, 1e-30f)), MIN_NORM);
    float s2 = dsinh(tn) / tn;
    float on = s2 * tn;
    float o0 = sqrtf(fmaxf(1.f + on * on, EPS));
    float* orow = out + (size_t)(r0 + row) * DDIM + sub * 8;
    #pragma unroll
    for (int i = 0; i < 8; ++i){
      bool c0 = (sub == 0 && i == 0);
      orow[i] = c0 ? o0 : s2 * t[i];
    }
  }
}

// ---------------------------------------------------------------------------
extern "C" void kernel_launch(void* const* d_in, const int* in_sizes, int n_in,
                              void* d_out, int out_size, void* d_ws, size_t ws_size,
                              hipStream_t stream){
  const float* x    = (const float*)d_in[0];   // 8192*64
  const float* adj  = (const float*)d_in[1];   // 8192*8192
  const float* W    = (const float*)d_in[2];   // 64*64
  const float* bias = (const float*)d_in[3];   // 64
  float* out = (float*)d_out;                  // 8192*64

  float*  ub  = (float*)d_ws;                            // 64 floats (pad 256)
  __bf16* L2t = (__bf16*)((char*)d_ws + 1024);           // 64 x 8192 bf16 (1MB)

  bias_prep<<<1, 64, 0, stream>>>(bias, ub);
  pre_kernel<<<NROWS / 16, 128, 0, stream>>>(x, W, ub, L2t);
  agg_kernel<<<NROWS / 16, 128, 0, stream>>>(adj, L2t, out);
}
